// Qwen35TextAttention_12146167513342
// MI455X (gfx1250) — matrix operand order repaired and checked by reading, not compiled, not run on hardware
//
#include <hip/hip_runtime.h>

typedef __attribute__((ext_vector_type(16))) _Float16 v16h;
typedef __attribute__((ext_vector_type(8)))  float    v8f;

#define T_TOK     2048
#define HIDDEN    2048
#define NUM_HEADS 16
#define NUM_KV    8
#define HEAD_DIM  128
#define HALF_DIM  64
#define QKV_COLS  ((NUM_HEADS + 2 * NUM_KV) * HEAD_DIM)   // 4096
#define Q_STRIDE  (NUM_HEADS * HEAD_DIM)                  // 2048
#define KV_STRIDE (NUM_KV * HEAD_DIM)                     // 1024
#define ATTN_SCALE 0.08838834764831845f
#define RMS_EPS    1e-6f

// ---------------------------------------------------------------------------
// mRoPE cos/sin tables. For HEAD_DIM=128, MROPE_SECTION=[21,21,22] reduces to
// freq-dim = i % 3 for i in [0,64).
// ---------------------------------------------------------------------------
__global__ void __launch_bounds__(64)
rope_tables_kernel(const int* __restrict__ pos,
                   float* __restrict__ cosT, float* __restrict__ sinT) {
    int t = blockIdx.x;
    int i = threadIdx.x;                      // 0..63
    int dim = i % 3;
    float p = (float)pos[dim * T_TOK + t];
    float inv = __powf(1.0e6f, -(float)i / 64.0f);
    float f = p * inv;
    float c = __cosf(f);
    float s = __sinf(f);
    cosT[t * HEAD_DIM + i]            = c;
    cosT[t * HEAD_DIM + HALF_DIM + i] = c;
    sinT[t * HEAD_DIM + i]            = s;
    sinT[t * HEAD_DIM + HALF_DIM + i] = s;
}

// ---------------------------------------------------------------------------
// Elementwise f32 -> f16 (vectorized float4 -> 4 halves).
// ---------------------------------------------------------------------------
__global__ void __launch_bounds__(256)
cvt_f16_kernel(const float* __restrict__ in, _Float16* __restrict__ out, int n4) {
    int i = blockIdx.x * 256 + threadIdx.x;
    if (i >= n4) return;
    const float4 v = ((const float4*)in)[i];
    _Float16 h0 = (_Float16)v.x, h1 = (_Float16)v.y;
    _Float16 h2 = (_Float16)v.z, h3 = (_Float16)v.w;
    union { _Float16 h[4]; short2 s2[2]; } u;
    u.h[0] = h0; u.h[1] = h1; u.h[2] = h2; u.h[3] = h3;
    ((short2*)out)[i * 2 + 0] = u.s2[0];
    ((short2*)out)[i * 2 + 1] = u.s2[1];
}

// ---------------------------------------------------------------------------
// Tiled transpose with convert: in (rows x cols, f32) -> out (cols x rows, f16)
// ---------------------------------------------------------------------------
__global__ void __launch_bounds__(256)
transpose_cvt_kernel(const float* __restrict__ in, _Float16* __restrict__ out,
                     int rows, int cols) {
    __shared__ _Float16 tile[32][33];
    int c0 = blockIdx.x * 32, r0 = blockIdx.y * 32;
    int tx = threadIdx.x, ty = threadIdx.y;      // 32 x 8
#pragma unroll
    for (int i = 0; i < 32; i += 8)
        tile[ty + i][tx] = (_Float16)in[(size_t)(r0 + ty + i) * cols + c0 + tx];
    __syncthreads();
#pragma unroll
    for (int i = 0; i < 32; i += 8)
        out[(size_t)(c0 + ty + i) * rows + r0 + tx] = tile[tx][ty + i];
}

// ---------------------------------------------------------------------------
// Tiled transpose f16 -> f16: in (rows x cols) -> out (cols x rows)
// ---------------------------------------------------------------------------
__global__ void __launch_bounds__(256)
transpose_f16_kernel(const _Float16* __restrict__ in, _Float16* __restrict__ out,
                     int rows, int cols) {
    __shared__ _Float16 tile[32][33];
    int c0 = blockIdx.x * 32, r0 = blockIdx.y * 32;
    int tx = threadIdx.x, ty = threadIdx.y;
#pragma unroll
    for (int i = 0; i < 32; i += 8)
        tile[ty + i][tx] = in[(size_t)(r0 + ty + i) * cols + c0 + tx];
    __syncthreads();
#pragma unroll
    for (int i = 0; i < 32; i += 8)
        out[(size_t)(c0 + ty + i) * rows + r0 + tx] = tile[tx][ty + i];
}

// ---------------------------------------------------------------------------
// WMMA fragment loaders, all-contiguous f16 (wave32 layouts, ISA 7.12.2).
// A (16x32): lane m = lane&15, K base = (lane>=16)?16:0, e -> K=base+e
//            => 32 contiguous bytes.
// B (32x16) from transposed (NxK) storage: lane n = lane&15,
//            K base = (lane>=16)?16:0, e -> K=base+e => 32 contiguous bytes.
// ---------------------------------------------------------------------------
__device__ __forceinline__ v16h load_a_f16(const _Float16* __restrict__ A,
                                           int m0, int k0, int ldk, int lane) {
    const _Float16* row = A + (size_t)(m0 + (lane & 15)) * ldk + k0 + ((lane >> 4) << 4);
    __builtin_prefetch((const void*)(row + 32), 0, 3);   // next k-tile
    v16h a;
#pragma unroll
    for (int e = 0; e < 16; ++e)
        a[e] = row[e];
    return a;
}

__device__ __forceinline__ v16h load_bt_f16(const _Float16* __restrict__ BT,
                                            int n0, int k0, int ldk, int lane) {
    const _Float16* row = BT + (size_t)(n0 + (lane & 15)) * ldk + k0 + ((lane >> 4) << 4);
    v16h b;
#pragma unroll
    for (int e = 0; e < 16; ++e)
        b[e] = row[e];
    return b;
}

// ---------------------------------------------------------------------------
// GEMM: C(MxN,f32) = A(MxK,f16) * BT(NxK,f16)^T, f32 accumulate.
// Wave computes 32x64 (2 A-frags x 4 B-frags -> 8 WMMAs per k-step).
// Block = 4 waves stacked in M = 128x64 tile.
// ---------------------------------------------------------------------------
__global__ void __launch_bounds__(128)
gemm_wmma_f16t(const _Float16* __restrict__ A, const _Float16* __restrict__ BT,
               float* __restrict__ C, int K, int ldc) {
    int lane = threadIdx.x & 31;
    int wave = threadIdx.x >> 5;
    int m0 = blockIdx.y * 128 + wave * 32;
    int n0 = blockIdx.x * 64;

    v8f acc0[4], acc1[4];
#pragma unroll
    for (int t = 0; t < 4; ++t)
#pragma unroll
        for (int r = 0; r < 8; ++r) { acc0[t][r] = 0.0f; acc1[t][r] = 0.0f; }

    for (int k0 = 0; k0 < K; k0 += 32) {
        v16h a0 = load_a_f16(A, m0,      k0, K, lane);
        v16h a1 = load_a_f16(A, m0 + 16, k0, K, lane);
#pragma unroll
        for (int t = 0; t < 4; ++t) {
            v16h b = load_bt_f16(BT, n0 + 16 * t, k0, K, lane);
            acc0[t] = __builtin_amdgcn_wmma_f32_16x16x32_f16(false, a0, false, b, (short)0, acc0[t], false, false);
            acc1[t] = __builtin_amdgcn_wmma_f32_16x16x32_f16(false, a1, false, b, (short)0, acc1[t], false, false);
        }
    }

    int rbase = m0 + ((lane >> 4) << 3);
    int cbase = n0 + (lane & 15);
#pragma unroll
    for (int t = 0; t < 4; ++t)
#pragma unroll
        for (int r = 0; r < 8; ++r) {
            C[(size_t)(rbase + r)      * ldc + cbase + 16 * t] = acc0[t][r];
            C[(size_t)(rbase + r + 16) * ldc + cbase + 16 * t] = acc1[t][r];
        }
}

// ---------------------------------------------------------------------------
// RMS-norm (head_dim) + RoPE on q/k; v converted to f16 (row-major).
// ---------------------------------------------------------------------------
__global__ void __launch_bounds__(128)
norm_rope_kernel(const float* __restrict__ qkv,
                 const float* __restrict__ cosT, const float* __restrict__ sinT,
                 const float* __restrict__ qw, const float* __restrict__ kw,
                 _Float16* __restrict__ qh, _Float16* __restrict__ kh,
                 _Float16* __restrict__ vh) {
    int t = blockIdx.x;
    int slot = blockIdx.y;                 // uniform per block
    int d = threadIdx.x;                   // 0..127
    float x = qkv[(size_t)t * QKV_COLS + slot * HEAD_DIM + d];

    if (slot >= NUM_HEADS + NUM_KV) {      // V path
        vh[(size_t)t * KV_STRIDE + (slot - NUM_HEADS - NUM_KV) * HEAD_DIM + d] = (_Float16)x;
        return;
    }

    __shared__ float red[HEAD_DIM];
    __shared__ float sn[HEAD_DIM];
    red[d] = x * x;
    __syncthreads();
#pragma unroll
    for (int s = 64; s > 0; s >>= 1) {
        if (d < s) red[d] += red[d + s];
        __syncthreads();
    }
    float rinv = rsqrtf(red[0] * (1.0f / HEAD_DIM) + RMS_EPS);
    bool isq = slot < NUM_HEADS;
    float xn = x * rinv * (isq ? qw[d] : kw[d]);
    sn[d] = xn;
    __syncthreads();
    float rot = (d < HALF_DIM) ? -sn[d + HALF_DIM] : sn[d - HALF_DIM];
    float out = xn * cosT[t * HEAD_DIM + d] + rot * sinT[t * HEAD_DIM + d];
    if (isq) qh[(size_t)t * Q_STRIDE + slot * HEAD_DIM + d] = (_Float16)out;
    else     kh[(size_t)t * KV_STRIDE + (slot - NUM_HEADS) * HEAD_DIM + d] = (_Float16)out;
}

// ---------------------------------------------------------------------------
// Flash attention: one wave per (head, 16-query block); 4 waves per block.
// K-fragments contiguous (head_dim fastest); V read from transposed layout
// vT[kv_head*128 + dim][token] so PV B-fragments are contiguous too.
// Output written directly as f16 for the output projection GEMM.
// ---------------------------------------------------------------------------
__global__ void __launch_bounds__(128)
attn_kernel(const _Float16* __restrict__ qh, const _Float16* __restrict__ kh,
            const _Float16* __restrict__ vT, _Float16* __restrict__ attnH) {
    __shared__ _Float16 plds[4][16][40];   // per-wave 16x32 P tile (+pad)

    int lane = threadIdx.x & 31;
    int wave = threadIdx.x >> 5;
    int wid = blockIdx.x * 4 + wave;       // 0..2047
    int h   = wid >> 7;
    int q0  = (wid & 127) * 16;
    int kvh = h >> 1;                      // GQA: 2 q-heads per kv-head
    int l15 = lane & 15;
    int hi8  = ((lane >> 4) << 3);
    int hi16 = ((lane >> 4) << 4);

    // Q fragments: 4 chunks of 32 head-dims, A layout (contiguous chunks)
    v16h qf[4];
#pragma unroll
    for (int c = 0; c < 4; ++c) {
        const _Float16* qrow = qh + (size_t)(q0 + l15) * Q_STRIDE + h * HEAD_DIM + c * 32 + hi16;
#pragma unroll
        for (int e = 0; e < 16; ++e) qf[c][e] = qrow[e];
    }

    v8f o[8];
    float mrow[8], lrow[8];
#pragma unroll
    for (int t = 0; t < 8; ++t)
#pragma unroll
        for (int r = 0; r < 8; ++r) o[t][r] = 0.0f;
#pragma unroll
    for (int r = 0; r < 8; ++r) { mrow[r] = -1e30f; lrow[r] = 0.0f; }

    int kb_last = (q0 + 15) & ~31;
    for (int kb = 0; kb <= kb_last; kb += 32) {
        // ---- S = Q * K^T (two 16-key N-tiles) ----
        v8f s0 = {}, s1 = {};
#pragma unroll
        for (int c = 0; c < 4; ++c) {
            const _Float16* kc0 = kh + (size_t)(kb + l15) * KV_STRIDE + kvh * HEAD_DIM + c * 32 + hi16;
            const _Float16* kc1 = kc0 + (size_t)16 * KV_STRIDE;
            v16h b0, b1;
#pragma unroll
            for (int e = 0; e < 16; ++e) { b0[e] = kc0[e]; b1[e] = kc1[e]; }
            s0 = __builtin_amdgcn_wmma_f32_16x16x32_f16(false, qf[c], false, b0, (short)0, s0, false, false);
            s1 = __builtin_amdgcn_wmma_f32_16x16x32_f16(false, qf[c], false, b1, (short)0, s1, false, false);
        }

        // ---- causal mask + online softmax (row spans 16 lanes of a half) ----
        int key0 = kb + l15;
        int key1 = key0 + 16;
        int rowb = q0 + hi8;
        float p0[8], p1[8], alpha[8];
#pragma unroll
        for (int r = 0; r < 8; ++r) {
            int qrow = rowb + r;
            float a0 = (key0 <= qrow) ? s0[r] * ATTN_SCALE : -1e30f;
            float a1 = (key1 <= qrow) ? s1[r] * ATTN_SCALE : -1e30f;
            float mx = fmaxf(a0, a1);
#pragma unroll
            for (int off = 8; off > 0; off >>= 1)
                mx = fmaxf(mx, __shfl_xor(mx, off, 32));
            float mnew = fmaxf(mrow[r], mx);
            float e0 = __expf(a0 - mnew);
            float e1 = __expf(a1 - mnew);
            float rs = e0 + e1;
#pragma unroll
            for (int off = 8; off > 0; off >>= 1)
                rs += __shfl_xor(rs, off, 32);
            alpha[r] = __expf(mrow[r] - mnew);
            lrow[r] = lrow[r] * alpha[r] + rs;
            mrow[r] = mnew;
            p0[r] = e0;
            p1[r] = e1;
        }
#pragma unroll
        for (int t = 0; t < 8; ++t)
#pragma unroll
            for (int r = 0; r < 8; ++r) o[t][r] *= alpha[r];

        // ---- transpose P (C layout -> A layout) through LDS ----
#pragma unroll
        for (int r = 0; r < 8; ++r) {
            plds[wave][hi8 + r][l15]      = (_Float16)p0[r];
            plds[wave][hi8 + r][16 + l15] = (_Float16)p1[r];
        }
        asm volatile("s_wait_dscnt 0" ::: "memory");  // cross-lane LDS RAW
        v16h pa;
        {
            const _Float16* prow = &plds[wave][l15][hi16];
#pragma unroll
            for (int e = 0; e < 16; ++e) pa[e] = prow[e];
        }

        // ---- O += P * V (8 head-dim N-tiles, K = 32 keys, contiguous) ----
#pragma unroll
        for (int t = 0; t < 8; ++t) {
            const _Float16* vrow = vT + (size_t)(kvh * HEAD_DIM + t * 16 + l15) * T_TOK + kb + hi16;
            v16h bv;
#pragma unroll
            for (int e = 0; e < 16; ++e) bv[e] = vrow[e];
            o[t] = __builtin_amdgcn_wmma_f32_16x16x32_f16(false, pa, false, bv, (short)0, o[t], false, false);
        }
    }

    // ---- normalize + store f16 [T, 16*128] ----
    int rowb = q0 + hi8;
#pragma unroll
    for (int t = 0; t < 8; ++t) {
        int col = h * HEAD_DIM + t * 16 + l15;
#pragma unroll
        for (int r = 0; r < 8; ++r)
            attnH[(size_t)(rowb + r) * Q_STRIDE + col] = (_Float16)(o[t][r] / lrow[r]);
    }
}

// ---------------------------------------------------------------------------
extern "C" void kernel_launch(void* const* d_in, const int* in_sizes, int n_in,
                              void* d_out, int out_size, void* d_ws, size_t ws_size,
                              hipStream_t stream) {
    (void)in_sizes; (void)n_in; (void)out_size; (void)ws_size;
    const int*   positions = (const int*)d_in[0];
    const float* hidden    = (const float*)d_in[1];
    const float* w_qkv     = (const float*)d_in[2];
    const float* w_o       = (const float*)d_in[3];
    const float* q_norm_w  = (const float*)d_in[4];
    const float* k_norm_w  = (const float*)d_in[5];
    float* out = (float*)d_out;

    // workspace layout
    float* cosT = (float*)d_ws;                                    // T*128 f32
    float* sinT = cosT + (size_t)T_TOK * HEAD_DIM;                 // T*128 f32
    float* qkv  = sinT + (size_t)T_TOK * HEAD_DIM;                 // T*4096 f32
    _Float16* hiddenH = (_Float16*)(qkv + (size_t)T_TOK * QKV_COLS);
    _Float16* wqkvT   = hiddenH + (size_t)T_TOK * HIDDEN;          // 4096x2048
    _Float16* woT     = wqkvT + (size_t)QKV_COLS * HIDDEN;         // 2048x2048
    _Float16* qh      = woT + (size_t)HIDDEN * HIDDEN;             // T*2048
    _Float16* kh      = qh + (size_t)T_TOK * Q_STRIDE;             // T*1024
    _Float16* vh      = kh + (size_t)T_TOK * KV_STRIDE;            // T*1024
    _Float16* vTr     = vh + (size_t)T_TOK * KV_STRIDE;            // 1024*T
    _Float16* attnH   = vTr + (size_t)KV_STRIDE * T_TOK;           // T*2048

    rope_tables_kernel<<<T_TOK, 64, 0, stream>>>(positions, cosT, sinT);

    cvt_f16_kernel<<<(T_TOK * HIDDEN / 4 + 255) / 256, 256, 0, stream>>>(
        hidden, hiddenH, T_TOK * HIDDEN / 4);

    transpose_cvt_kernel<<<dim3(QKV_COLS / 32, HIDDEN / 32), dim3(32, 8), 0, stream>>>(
        w_qkv, wqkvT, HIDDEN, QKV_COLS);
    transpose_cvt_kernel<<<dim3(HIDDEN / 32, HIDDEN / 32), dim3(32, 8), 0, stream>>>(
        w_o, woT, HIDDEN, HIDDEN);

    gemm_wmma_f16t<<<dim3(QKV_COLS / 64, T_TOK / 128), 128, 0, stream>>>(
        hiddenH, wqkvT, qkv, HIDDEN, QKV_COLS);

    norm_rope_kernel<<<dim3(T_TOK, NUM_HEADS + 2 * NUM_KV), 128, 0, stream>>>(
        qkv, cosT, sinT, q_norm_w, k_norm_w, qh, kh, vh);

    transpose_f16_kernel<<<dim3(KV_STRIDE / 32, T_TOK / 32), dim3(32, 8), 0, stream>>>(
        vh, vTr, T_TOK, KV_STRIDE);

    attn_kernel<<<(NUM_HEADS * (T_TOK / 16)) / 4, 128, 0, stream>>>(qh, kh, vTr, attnH);

    gemm_wmma_f16t<<<dim3(HIDDEN / 64, T_TOK / 128), 128, 0, stream>>>(
        attnH, woT, out, HIDDEN, HIDDEN);
}
